// OriDis_TransformerEncoder_39900246180063
// MI455X (gfx1250) — matrix operand order repaired and checked by reading, not compiled, not run on hardware
//
#include <hip/hip_runtime.h>
#include <hip/hip_bf16.h>

// ---------------------------------------------------------------------------
// Transformer encoder layer forward for MI455X (gfx1250, wave32, WMMA).
// B=2, N=2048, E=1024, H=16, D=64, FF=4096.  ~137 GFLOP, ~50MB working set
// (fits in 192MB L2) -> matrix-core bound.  All GEMMs use
// v_wmma_f32_16x16x32_f16 with f32 accumulation.  GEMM LDS tiles are filled
// with async global->LDS DMA (ASYNCcnt) and double-buffered when available.
// ---------------------------------------------------------------------------

typedef _Float16 v16h __attribute__((ext_vector_type(16)));
typedef _Float16 v8h  __attribute__((ext_vector_type(8)));
typedef float    v8f  __attribute__((ext_vector_type(8)));
typedef int      v4i  __attribute__((vector_size(4 * sizeof(int))));

#define BDIM 2
#define NSEQ 2048
#define EMB  1024
#define NH   16
#define HD   64
#define FFD  4096
#define ROWS (BDIM * NSEQ)   // 4096 tokens

// ---- CDNA5 async global->LDS copy (guarded; falls back to sync path) ------
#if defined(__has_builtin)
#  if __has_builtin(__builtin_amdgcn_global_load_async_to_lds_b128) && \
      __has_builtin(__builtin_amdgcn_s_wait_asynccnt)
#    define HAVE_ASYNC_COPY 1
#  endif
#endif
#ifndef HAVE_ASYNC_COPY
#  define HAVE_ASYNC_COPY 0
#endif

typedef __attribute__((address_space(1))) v4i gv4i;   // global 128-bit chunk
typedef __attribute__((address_space(3))) v4i lv4i;   // LDS    128-bit chunk

__device__ __forceinline__ void async_copy16(const _Float16* g, _Float16* l) {
#if HAVE_ASYNC_COPY
  __builtin_amdgcn_global_load_async_to_lds_b128((gv4i*)g, (lv4i*)l, 0, 0);
#else
  *(v8h*)l = *(const v8h*)g;
#endif
}

__device__ __forceinline__ void async_wait0() {
#if HAVE_ASYNC_COPY
  __builtin_amdgcn_s_wait_asynccnt(0);
#endif
}

__device__ __forceinline__ v16h cat8(v8h lo, v8h hi) {
  v16h r;
#pragma unroll
  for (int i = 0; i < 8; ++i) { r[i] = lo[i]; r[i + 8] = hi[i]; }
  return r;
}

__device__ __forceinline__ v8f wmma_f16(v16h a, v16h b, v8f c) {
  // D = A(16x32 f16) * B(32x16 f16) + C(16x16 f32)
  return __builtin_amdgcn_wmma_f32_16x16x32_f16(false, a, false, b,
                                                (short)0, c, false, false);
}

// A-matrix fragment (16x32 f16) from row-major halves with given stride.
// lane L: row = L%16, half-chunks at c0 = 16*(L/16) and c0+8.
__device__ __forceinline__ v16h loadA(const _Float16* base, int stride, int lane) {
  int r  = lane & 15;
  int c0 = (lane >> 4) * 16;
  v8h lo = *(const v8h*)(base + (size_t)r * stride + c0);
  v8h hi = *(const v8h*)(base + (size_t)r * stride + c0 + 8);
  return cat8(lo, hi);
}

// ---------------------------------------------------------------------------
// Weight transpose + f32->f16:  W[K][Nc] -> Wt[Nc][K]  (k contiguous so that
// WMMA B-fragments are contiguous 32B loads per lane).
// ---------------------------------------------------------------------------
__global__ __launch_bounds__(256) void k_tconv(const float* __restrict__ W,
                                               _Float16* __restrict__ Wt,
                                               int K, int Nc) {
  __shared__ float tile[32][33];
  int tx = threadIdx.x & 31, ty = threadIdx.x >> 5;   // 32 x 8
  int bx = blockIdx.x, by = blockIdx.y;
#pragma unroll
  for (int i = 0; i < 4; ++i) {
    int kk = by * 32 + ty + i * 8;
    tile[ty + i * 8][tx] = W[(size_t)kk * Nc + bx * 32 + tx];
  }
  __syncthreads();
#pragma unroll
  for (int i = 0; i < 4; ++i) {
    int c = bx * 32 + ty + i * 8;
    Wt[(size_t)c * K + by * 32 + tx] = (_Float16)tile[tx][ty + i * 8];
  }
}

__global__ void k_concat3(const float* __restrict__ a, const float* __restrict__ b,
                          const float* __restrict__ c, float* __restrict__ out) {
  int i = threadIdx.x + blockIdx.x * blockDim.x;
  if (i < EMB)            out[i] = a[i];
  else if (i < 2 * EMB)   out[i] = b[i - EMB];
  else if (i < 3 * EMB)   out[i] = c[i - 2 * EMB];
}

// ---------------------------------------------------------------------------
// Fused input projection + LayerNorm.  One 256-thread block per token row.
// ---------------------------------------------------------------------------
__device__ __forceinline__ void blockReduce2(float& s, float& ss) {
  __shared__ float rs[2][8];
#pragma unroll
  for (int off = 1; off < 32; off <<= 1) {
    s  += __shfl_xor(s,  off, 32);
    ss += __shfl_xor(ss, off, 32);
  }
  int w = threadIdx.x >> 5;
  if ((threadIdx.x & 31) == 0) { rs[0][w] = s; rs[1][w] = ss; }
  __syncthreads();
  s  = rs[0][threadIdx.x & 7];
  ss = rs[1][threadIdx.x & 7];
#pragma unroll
  for (int off = 1; off < 8; off <<= 1) {
    s  += __shfl_xor(s,  off, 8);
    ss += __shfl_xor(ss, off, 8);
  }
}

__global__ __launch_bounds__(256) void k_inproj_ln(
    const float* __restrict__ x, const float* __restrict__ Win,
    const float* __restrict__ bin, const float* __restrict__ gamma,
    const float* __restrict__ beta, float* __restrict__ h,
    _Float16* __restrict__ hL) {
  int row = blockIdx.x, tid = threadIdx.x;
  float x0 = x[row * 2 + 0], x1 = x[row * 2 + 1];
  float v[4], s = 0.f, ss = 0.f;
#pragma unroll
  for (int i = 0; i < 4; ++i) {
    int c = tid + i * 256;
    v[i] = x0 * Win[c] + x1 * Win[EMB + c] + bin[c];
    h[(size_t)row * EMB + c] = v[i];
    s += v[i]; ss += v[i] * v[i];
  }
  blockReduce2(s, ss);
  float mu = s * (1.f / EMB);
  float var = ss * (1.f / EMB) - mu * mu;
  float rstd = rsqrtf(var + 1e-5f);
#pragma unroll
  for (int i = 0; i < 4; ++i) {
    int c = tid + i * 256;
    hL[(size_t)row * EMB + c] = (_Float16)((v[i] - mu) * rstd * gamma[c] + beta[c]);
  }
}

// Standalone LayerNorm (f32 in -> f16 out), one block per row.
__global__ __launch_bounds__(256) void k_ln(const float* __restrict__ in,
                                            const float* __restrict__ gamma,
                                            const float* __restrict__ beta,
                                            _Float16* __restrict__ out) {
  int row = blockIdx.x, tid = threadIdx.x;
  const float* p = in + (size_t)row * EMB;
  float v[4], s = 0.f, ss = 0.f;
#pragma unroll
  for (int i = 0; i < 4; ++i) {
    v[i] = p[tid + i * 256];
    s += v[i]; ss += v[i] * v[i];
  }
  blockReduce2(s, ss);
  float mu = s * (1.f / EMB);
  float var = ss * (1.f / EMB) - mu * mu;
  float rstd = rsqrtf(var + 1e-5f);
#pragma unroll
  for (int i = 0; i < 4; ++i) {
    int c = tid + i * 256;
    out[(size_t)row * EMB + c] = (_Float16)((v[i] - mu) * rstd * gamma[c] + beta[c]);
  }
}

// ---------------------------------------------------------------------------
// LDS-tiled WMMA GEMM:  C[M][Nc] = A[M][K](f16) * W[K][Nc] + bias, with W
// pre-transposed as Wt[Nc][K].  Block: 256 thr (8 waves), tile 128x128x32,
// double-buffered async global->LDS fills.  Wave tile 32x64 = 2x4 WMMA tiles.
// ---------------------------------------------------------------------------
enum { EPI_QKV = 0, EPI_RES_F32 = 1, EPI_GELU_F16 = 2 };

#define GBM 128
#define GBK 32
#define GLD 40   // padded LDS row stride (halves): 80B, kills phase conflicts

__device__ __forceinline__ void fill_tile(_Float16* As, _Float16* Bs,
                                          const _Float16* __restrict__ A,
                                          const _Float16* __restrict__ Wt,
                                          int bRow, int bCol, int K, int k0,
                                          int tid) {
#pragma unroll
  for (int it = 0; it < 2; ++it) {
    int idx = tid + it * 256;          // 0..511 -> 128 rows x 4 chunks of 8h
    int r   = idx >> 2;
    int c8  = (idx & 3) * 8;
    async_copy16(&A [(size_t)(bRow + r) * K + k0 + c8], &As[r * GLD + c8]);
    async_copy16(&Wt[(size_t)(bCol + r) * K + k0 + c8], &Bs[r * GLD + c8]);
  }
}

template <int EPI>
__global__ __launch_bounds__(256) void k_gemm(
    const _Float16* __restrict__ A, const _Float16* __restrict__ Wt,
    const float* __restrict__ bias, const float* __restrict__ resid,
    float* __restrict__ outF, _Float16* __restrict__ outH,
    _Float16* __restrict__ qb, _Float16* __restrict__ kb,
    _Float16* __restrict__ vTb, int M, int K, int Nc) {
  __shared__ _Float16 As[2][GBM * GLD];
  __shared__ _Float16 Bs[2][GBM * GLD];

  const int tid  = threadIdx.x;
  const int lane = tid & 31;
  const int wid  = tid >> 5;
  const int wr   = wid & 3;         // 4 row-blocks of 32
  const int wc   = wid >> 2;        // 2 col-blocks of 64
  const int bRow = (int)blockIdx.y * GBM;
  const int bCol = (int)blockIdx.x * GBM;
  const int cl   = lane & 15;
  const int hsel = lane >> 4;

  v8f zero = {};
  v8f acc[2][4];
#pragma unroll
  for (int mt = 0; mt < 2; ++mt)
#pragma unroll
    for (int nt = 0; nt < 4; ++nt) acc[mt][nt] = zero;

  int p = 0;
  fill_tile(As[0], Bs[0], A, Wt, bRow, bCol, K, 0, tid);

  for (int k0 = 0; k0 < K; k0 += GBK) {
    async_wait0();         // my DMA into buffer p is done
    __syncthreads();       // everyone's DMA done; prior reads of p^1 done
    if (k0 + GBK < K)      // prefetch next tile into the other buffer
      fill_tile(As[p ^ 1], Bs[p ^ 1], A, Wt, bRow, bCol, K, k0 + GBK, tid);

    v16h af[2], bf[4];
#pragma unroll
    for (int mt = 0; mt < 2; ++mt)
      af[mt] = loadA(&As[p][(wr * 32 + mt * 16) * GLD], GLD, lane);
#pragma unroll
    for (int nt = 0; nt < 4; ++nt) {
      const _Float16* q = &Bs[p][(wc * 64 + nt * 16 + cl) * GLD + hsel * 16];
      bf[nt] = cat8(*(const v8h*)q, *(const v8h*)(q + 8));
    }
#pragma unroll
    for (int mt = 0; mt < 2; ++mt)
#pragma unroll
      for (int nt = 0; nt < 4; ++nt)
        acc[mt][nt] = wmma_f16(af[mt], bf[nt], acc[mt][nt]);
    p ^= 1;
  }

  // Epilogue.  C layout: lane, VGPR j -> row = j + 8*(lane/16), col = lane%16.
#pragma unroll
  for (int mt = 0; mt < 2; ++mt)
#pragma unroll
    for (int nt = 0; nt < 4; ++nt)
#pragma unroll
      for (int j = 0; j < 8; ++j) {
        int row = bRow + wr * 32 + mt * 16 + j + 8 * hsel;
        int col = bCol + wc * 64 + nt * 16 + cl;
        float v = acc[mt][nt][j] + bias[col];
        if (EPI == EPI_QKV) {
          int which = col >> 10;          // 0=q 1=k 2=v
          int cc = col & 1023;
          int hh = cc >> 6, d = cc & 63;
          int b  = row >> 11, n = row & 2047;
          size_t bh = (size_t)(b * NH + hh);
          if (which == 0)
            qb [(bh * NSEQ + n) * HD + d] = (_Float16)v;
          else if (which == 1)
            kb [(bh * NSEQ + n) * HD + d] = (_Float16)v;
          else
            vTb[(bh * HD + d) * NSEQ + n] = (_Float16)v;   // V transposed
        } else if (EPI == EPI_RES_F32) {
          size_t o = (size_t)row * Nc + col;
          outF[o] = v + resid[o];
        } else {  // EPI_GELU_F16 (exact gelu)
          float g = 0.5f * v * (1.f + erff(v * 0.70710678118654752f));
          outH[(size_t)row * Nc + col] = (_Float16)g;
        }
      }
}

// ---------------------------------------------------------------------------
// Flash attention.  grid = (B*H, N/64); block = 128 (4 waves).  Each wave owns
// a 16-query tile and streams keys in blocks of 32 with online softmax.
// S = Q*K^T via 2 WMMAs per 16-key tile; P goes C-layout -> LDS -> A-layout;
// O += P*V via 4 WMMAs (V stored transposed: vT[bh][d][n]).
// softmax scale = 1/sqrt(E) = 1/32 (as in reference).
// ---------------------------------------------------------------------------
__global__ __launch_bounds__(128) void k_attn(
    const _Float16* __restrict__ q, const _Float16* __restrict__ k,
    const _Float16* __restrict__ vT, _Float16* __restrict__ out) {
  __shared__ _Float16 Pb[4][16 * 40];       // per-wave 16x32 P tile, stride 40
  const int lane = threadIdx.x & 31;
  const int w    = threadIdx.x >> 5;
  const int bh   = blockIdx.x;
  const int qBase = (int)blockIdx.y * 64 + w * 16;
  const int r    = lane & 15;
  const int hsel = lane >> 4;

  const _Float16* qp = q  + (size_t)bh * NSEQ * HD;
  const _Float16* kp = k  + (size_t)bh * NSEQ * HD;
  const _Float16* vp = vT + (size_t)bh * HD * NSEQ;

  // Q fragments for d-chunks [0,32) and [32,64).
  v16h qf[2];
#pragma unroll
  for (int kc = 0; kc < 2; ++kc) {
    const _Float16* base = qp + (size_t)(qBase + r) * HD + kc * 32 + hsel * 16;
    qf[kc] = cat8(*(const v8h*)base, *(const v8h*)(base + 8));
  }

  float m[8], l[8];
  v8f o[4];
  v8f zero = {};
#pragma unroll
  for (int j = 0; j < 8; ++j) { m[j] = -1e30f; l[j] = 0.f; }
#pragma unroll
  for (int n = 0; n < 4; ++n) o[n] = zero;

  const float scale = 0.03125f;  // 1/sqrt(1024)
  _Float16* pb = &Pb[w][0];

  for (int kbase = 0; kbase < NSEQ; kbase += 32) {
    // ---- S tiles (16 queries x 32 keys) ----
    v8f s[2];
#pragma unroll
    for (int t = 0; t < 2; ++t) {
      v8f c = zero;
#pragma unroll
      for (int kc = 0; kc < 2; ++kc) {
        // B frag: col = key, rows = 16 contiguous d values
        const _Float16* bp =
            kp + (size_t)(kbase + t * 16 + r) * HD + kc * 32 + hsel * 16;
        v16h bf = cat8(*(const v8h*)bp, *(const v8h*)(bp + 8));
        c = wmma_f16(qf[kc], bf, c);
      }
      s[t] = c;
    }
    // ---- online softmax stats (row = j + 8*hsel, spread over 16 lanes) ----
#pragma unroll
    for (int j = 0; j < 8; ++j) {
      float s0 = s[0][j] * scale, s1 = s[1][j] * scale;
      float mx = fmaxf(s0, s1);
#pragma unroll
      for (int off = 1; off < 16; off <<= 1) mx = fmaxf(mx, __shfl_xor(mx, off, 16));
      float mn   = fmaxf(m[j], mx);
      float corr = __expf(m[j] - mn);
      m[j] = mn;
      float p0 = __expf(s0 - mn), p1 = __expf(s1 - mn);
      s[0][j] = p0; s[1][j] = p1;
      float sum = p0 + p1;
#pragma unroll
      for (int off = 1; off < 16; off <<= 1) sum += __shfl_xor(sum, off, 16);
      l[j] = l[j] * corr + sum;
#pragma unroll
      for (int n = 0; n < 4; ++n) o[n][j] *= corr;
    }
    // ---- P: C-layout -> LDS -> A-layout fragment ----
#pragma unroll
    for (int t = 0; t < 2; ++t)
#pragma unroll
      for (int j = 0; j < 8; ++j)
        pb[(j + 8 * hsel) * 40 + t * 16 + r] = (_Float16)s[t][j];
    // same-wave LDS RAW: backend inserts s_wait_dscnt
    const _Float16* pa = pb + r * 40 + 16 * hsel;
    v16h pf = cat8(*(const v8h*)pa, *(const v8h*)(pa + 8));
    // ---- O += P * V ----
#pragma unroll
    for (int n = 0; n < 4; ++n) {
      const _Float16* vb = vp + (size_t)(n * 16 + r) * NSEQ + kbase + hsel * 16;
      v16h vf = cat8(*(const v8h*)vb, *(const v8h*)(vb + 8));
      o[n] = wmma_f16(pf, vf, o[n]);
    }
  }

  // ---- write normalized output to attnout[token][h*64+d] (f16) ----
  int b = bh >> 4, hh = bh & 15;
#pragma unroll
  for (int n = 0; n < 4; ++n)
#pragma unroll
    for (int j = 0; j < 8; ++j) {
      int row = (b << 11) + qBase + j + 8 * hsel;
      int col = (hh << 6) + n * 16 + r;
      out[(size_t)row * EMB + col] = (_Float16)(o[n][j] / l[j]);
    }
}

// ---------------------------------------------------------------------------
// Host-side orchestration.
// ---------------------------------------------------------------------------
static constexpr size_t MB = 1024ull * 1024ull;
static constexpr size_t OFF_H     = 0;          // f32 [4096][1024]   16MB
static constexpr size_t OFF_HL    = 16  * MB;   // f16 [4096][1024]    8MB
static constexpr size_t OFF_H2    = 24  * MB;   // f32 [4096][1024]   16MB
static constexpr size_t OFF_HL2   = 40  * MB;   // f16 [4096][1024]    8MB
static constexpr size_t OFF_Q     = 48  * MB;   // f16 [32][2048][64]  8MB
static constexpr size_t OFF_K     = 56  * MB;   // f16 [32][2048][64]  8MB
static constexpr size_t OFF_VT    = 64  * MB;   // f16 [32][64][2048]  8MB
static constexpr size_t OFF_AO    = 72  * MB;   // f16 [4096][1024]    8MB
static constexpr size_t OFF_FFN1  = 80  * MB;   // f16 [4096][4096]   32MB
static constexpr size_t OFF_WQKVT = 112 * MB;   // f16 [3072][1024]    6MB
static constexpr size_t OFF_WOT   = 118 * MB;   // f16 [1024][1024]    2MB
static constexpr size_t OFF_W1T   = 120 * MB;   // f16 [4096][1024]    8MB
static constexpr size_t OFF_W2T   = 128 * MB;   // f16 [1024][4096]    8MB
static constexpr size_t OFF_BQKV  = 136 * MB;   // f32 [3072]        12KB

extern "C" void kernel_launch(void* const* d_in, const int* in_sizes, int n_in,
                              void* d_out, int out_size, void* d_ws, size_t ws_size,
                              hipStream_t stream) {
  (void)in_sizes; (void)n_in; (void)out_size; (void)ws_size;
  const float* x     = (const float*)d_in[0];
  const float* Win   = (const float*)d_in[1];
  const float* bin   = (const float*)d_in[2];
  const float* gamma = (const float*)d_in[3];
  const float* beta  = (const float*)d_in[4];
  const float* Wq = (const float*)d_in[5];  const float* bq = (const float*)d_in[6];
  const float* Wk = (const float*)d_in[7];  const float* bk = (const float*)d_in[8];
  const float* Wv = (const float*)d_in[9];  const float* bv = (const float*)d_in[10];
  const float* Wo = (const float*)d_in[11]; const float* bo = (const float*)d_in[12];
  const float* W1 = (const float*)d_in[13]; const float* b1 = (const float*)d_in[14];
  const float* W2 = (const float*)d_in[15]; const float* b2 = (const float*)d_in[16];

  char* ws = (char*)d_ws;
  float*    h     = (float*)(ws + OFF_H);
  _Float16* hL    = (_Float16*)(ws + OFF_HL);
  float*    h2    = (float*)(ws + OFF_H2);
  _Float16* hL2   = (_Float16*)(ws + OFF_HL2);
  _Float16* qb    = (_Float16*)(ws + OFF_Q);
  _Float16* kbuf  = (_Float16*)(ws + OFF_K);
  _Float16* vTb   = (_Float16*)(ws + OFF_VT);
  _Float16* ao    = (_Float16*)(ws + OFF_AO);
  _Float16* ffn1  = (_Float16*)(ws + OFF_FFN1);
  _Float16* wqkvt = (_Float16*)(ws + OFF_WQKVT);
  _Float16* wot   = (_Float16*)(ws + OFF_WOT);
  _Float16* w1t   = (_Float16*)(ws + OFF_W1T);
  _Float16* w2t   = (_Float16*)(ws + OFF_W2T);
  float*    bqkv  = (float*)(ws + OFF_BQKV);

  // ---- one-time (per launch) weight transpose + f16 convert ----
  k_tconv<<<dim3(32, 32),  256, 0, stream>>>(Wq, wqkvt,                 EMB, EMB);
  k_tconv<<<dim3(32, 32),  256, 0, stream>>>(Wk, wqkvt + 1024 * 1024,   EMB, EMB);
  k_tconv<<<dim3(32, 32),  256, 0, stream>>>(Wv, wqkvt + 2 * 1024 * 1024, EMB, EMB);
  k_tconv<<<dim3(32, 32),  256, 0, stream>>>(Wo, wot,                   EMB, EMB);
  k_tconv<<<dim3(128, 32), 256, 0, stream>>>(W1, w1t,                   EMB, FFD);
  k_tconv<<<dim3(32, 128), 256, 0, stream>>>(W2, w2t,                   FFD, EMB);
  k_concat3<<<12, 256, 0, stream>>>(bq, bk, bv, bqkv);

  // ---- h = x@W_in + b_in ; hL = LN(h) ----
  k_inproj_ln<<<ROWS, 256, 0, stream>>>(x, Win, bin, gamma, beta, h, hL);

  // ---- fused QKV projection (Nc = 3072) with head scatter ----
  k_gemm<EPI_QKV><<<dim3(24, 32), 256, 0, stream>>>(
      hL, wqkvt, bqkv, nullptr, nullptr, nullptr, qb, kbuf, vTb,
      ROWS, EMB, 3 * EMB);

  // ---- flash attention ----
  k_attn<<<dim3(BDIM * NH, NSEQ / 64), 128, 0, stream>>>(qb, kbuf, vTb, ao);

  // ---- output projection + residual: h2 = h + (ao@Wo + bo) ----
  k_gemm<EPI_RES_F32><<<dim3(8, 32), 256, 0, stream>>>(
      ao, wot, bo, h, h2, nullptr, nullptr, nullptr, nullptr,
      ROWS, EMB, EMB);

  // ---- hL2 = LN(h2) ----
  k_ln<<<ROWS, 256, 0, stream>>>(h2, gamma, beta, hL2);

  // ---- FFN1: gelu(hL2@W1 + b1) -> f16 ----
  k_gemm<EPI_GELU_F16><<<dim3(32, 32), 256, 0, stream>>>(
      hL2, w1t, b1, nullptr, nullptr, ffn1, nullptr, nullptr, nullptr,
      ROWS, EMB, FFD);

  // ---- FFN2 + residual: out = h2 + (ffn1@W2 + b2) ----
  k_gemm<EPI_RES_F32><<<dim3(8, 32), 256, 0, stream>>>(
      ffn1, w2t, b2, h2, (float*)d_out, nullptr, nullptr, nullptr, nullptr,
      ROWS, FFD, EMB);
}
